// IndustryGNNPath_10771777978573
// MI455X (gfx1250) — matrix operand order repaired and checked by reading, not compiled, not run on hardware
//
#include <hip/hip_runtime.h>
#include <hip/hip_bf16.h>

typedef __bf16 bf16;
typedef __bf16  v16bf __attribute__((ext_vector_type(16)));
typedef __bf16  v8bf  __attribute__((ext_vector_type(8)));
typedef float   v8f   __attribute__((ext_vector_type(8)));

#define NN 20000
#define EE 250000
#define BB 128
#define HH 256
#define RR 64
#define BOND 13
#define INF_ 47
#define CUTOFF 10.0f

// ---------------------------------------------------------------- helpers
__device__ __forceinline__ float actf(float x, int act) {
  switch (act) {
    case 1: return x / (1.f + __expf(-x));                                  // silu
    case 2: { float c = 0.7978845608028654f * (x + 0.044715f * x * x * x);  // gelu(tanh)
              return 0.5f * x * (1.f + tanhf(c)); }
    case 3: return fmaxf(x, 0.f);                                           // relu
    case 4: return 1.f / (1.f + __expf(-x));                                // sigmoid
  }
  return x;
}
__device__ __forceinline__ void atomAdd(float* p, float v) {
  __hip_atomic_fetch_add(p, v, __ATOMIC_RELAXED, __HIP_MEMORY_SCOPE_AGENT);
}
__device__ __forceinline__ void atomicMaxF(float* addr, float val) {
  if (val >= 0.f) atomicMax((int*)addr, __float_as_int(val));
  else            atomicMin((unsigned int*)addr, __float_as_uint(val));
}
__device__ __forceinline__ v16bf cat16(v8bf lo, v8bf hi) {
  return __builtin_shufflevector(lo, hi, 0, 1, 2, 3, 4, 5, 6, 7,
                                 8, 9, 10, 11, 12, 13, 14, 15);
}

// ---------------------------------------------------------------- WMMA GEMM
// C[M,Nout] = act( (accum? C:0) + A[M,K]bf16 (opt row-gather) @ Wt[Nout,Kpad]^T + bias )
// Requirements (guaranteed by host): M % 16 == 0, Nout % 64 == 0, K % 32 == 0,
// A rows and Wt rows 16B-aligned, K-padding zero-filled. One wave -> 16x64 tile.
__global__ __launch_bounds__(32) void k_gemm(
    const bf16* __restrict__ A, long lda, const int* __restrict__ gidx,
    const bf16* __restrict__ Wt, long ldw, const float* __restrict__ bias,
    float* __restrict__ Cf, bf16* __restrict__ Cb, long ldc,
    int Nout, int K, int act, int accum) {
  const int row0 = blockIdx.x * 16;
  const int nt0  = blockIdx.y * 64;
  const int l  = threadIdx.x;
  const int lm = l & 15, lh = l >> 4;
  v8f c0 = {}, c1 = {}, c2 = {}, c3 = {};

  const int  Mi   = row0 + lm;
  const long arow = (gidx ? (long)gidx[Mi] : (long)Mi) * lda;
  const bf16* __restrict__ Ap = A + arow + lh * 16;
  const bf16* __restrict__ W0 = Wt + (long)(nt0 + lm) * ldw + lh * 16;
  const bf16* __restrict__ W1 = W0 + 16 * ldw;
  const bf16* __restrict__ W2 = W0 + 32 * ldw;
  const bf16* __restrict__ W3 = W0 + 48 * ldw;

  for (int kk = 0; kk < K; kk += 32) {
    __builtin_prefetch(Ap + kk + 64, 0, 0);
    v16bf a = cat16(*(const v8bf*)(Ap + kk), *(const v8bf*)(Ap + kk + 8));
    v16bf b0 = cat16(*(const v8bf*)(W0 + kk), *(const v8bf*)(W0 + kk + 8));
    v16bf b1 = cat16(*(const v8bf*)(W1 + kk), *(const v8bf*)(W1 + kk + 8));
    v16bf b2 = cat16(*(const v8bf*)(W2 + kk), *(const v8bf*)(W2 + kk + 8));
    v16bf b3 = cat16(*(const v8bf*)(W3 + kk), *(const v8bf*)(W3 + kk + 8));
    c0 = __builtin_amdgcn_wmma_f32_16x16x32_bf16(false, a, false, b0, (short)0, c0, false, false);
    c1 = __builtin_amdgcn_wmma_f32_16x16x32_bf16(false, a, false, b1, (short)0, c1, false, false);
    c2 = __builtin_amdgcn_wmma_f32_16x16x32_bf16(false, a, false, b2, (short)0, c2, false, false);
    c3 = __builtin_amdgcn_wmma_f32_16x16x32_bf16(false, a, false, b3, (short)0, c3, false, false);
  }
#pragma unroll
  for (int t = 0; t < 4; t++) {
    v8f cc = (t == 0) ? c0 : (t == 1) ? c1 : (t == 2) ? c2 : c3;
    const int col = nt0 + t * 16 + lm;
    const float bv = bias ? bias[col] : 0.f;
#pragma unroll
    for (int v = 0; v < 8; v++) {
      const long o = (long)(row0 + v + lh * 8) * ldc + col;
      float x = cc[v];
      if (accum) x += Cf[o];
      x = actf(x + bv, act);
      if (Cf) Cf[o] = x;
      if (Cb) Cb[o] = (bf16)x;
    }
  }
}

// ---------------------------------------------------------------- misc kernels
// w[K,N] f32 -> wt[N,Kpad] bf16, zero-padded for k in [K,Kpad)
__global__ void k_wt(const float* w, bf16* wt, int K, int N, int Kpad) {
  long t = (long)blockIdx.x * 256 + threadIdx.x;
  if (t >= (long)N * Kpad) return;
  int n = t / Kpad, k = t % Kpad;
  wt[t] = (k < K) ? (bf16)w[(long)k * N + n] : (bf16)0.f;
}
__global__ void k_fill(float* p, float v, long n) {
  long t = (long)blockIdx.x * 256 + threadIdx.x;
  if (t < n) p[t] = v;
}
// f32 [M,Dsrc] -> bf16 [M, ldd] writing Dpad cols (zero pad for c in [Dsrc,Dpad))
__global__ void k_f2b(const float* s, bf16* d, long M, int Dsrc, int Dpad, long ldd) {
  long t = (long)blockIdx.x * 256 + threadIdx.x;
  if (t >= M * Dpad) return;
  long r = t / Dpad; int c = t % Dpad;
  d[r * ldd + c] = (c < Dsrc) ? (bf16)s[r * Dsrc + c] : (bf16)0.f;
}
__global__ void k_ln(const float* x, const float* res, const float* g, const float* b,
                     float* outf, bf16* outb, long ldb, int D, long M) {
  long row = blockIdx.x; if (row >= M) return;
  int tid = threadIdx.x;
  __shared__ float sd[256];
  int nj = D >> 8;                     // D = 256 or 512
  const float* xr = x + row * D;
  const float* rr = res ? res + row * D : nullptr;
  float v[2] = {0.f, 0.f}, loc = 0.f;
  for (int j = 0; j < nj; j++) { int c = tid + j * 256; float t = xr[c]; if (rr) t += rr[c]; v[j] = t; loc += t; }
  sd[tid] = loc; __syncthreads();
  for (int s = 128; s > 0; s >>= 1) { if (tid < s) sd[tid] += sd[tid + s]; __syncthreads(); }
  float mean = sd[0] / D; __syncthreads();
  loc = 0.f;
  for (int j = 0; j < nj; j++) { float dd = v[j] - mean; loc += dd * dd; }
  sd[tid] = loc; __syncthreads();
  for (int s = 128; s > 0; s >>= 1) { if (tid < s) sd[tid] += sd[tid + s]; __syncthreads(); }
  float rstd = rsqrtf(sd[0] / D + 1e-5f);
  for (int j = 0; j < nj; j++) {
    int c = tid + j * 256;
    float y = g[c] * (v[j] - mean) * rstd + b[c];
    if (outf) outf[row * D + c] = y;
    if (outb) outb[row * ldb + c] = (bf16)y;
  }
}
__global__ __launch_bounds__(32) void k_rowdot(const float* A, int D, const float* w,
                                               const float* b, float* out, long M, int act) {
  long row = blockIdx.x; if (row >= M) return;
  int l = threadIdx.x;
  float acc = 0.f;
  for (int i = l; i < D; i += 32) acc += A[row * D + i] * w[i];
  for (int s = 16; s > 0; s >>= 1) acc += __shfl_down(acc, s, 32);
  if (l == 0) out[row] = actf(acc + (b ? b[0] : 0.f), act);
}
__global__ void k_edge_geom(const float* pos, const int* row, const int* col,
                            float* diff, float* dist, int E) {
  int e = blockIdx.x * 256 + threadIdx.x; if (e >= E) return;
  int r = row[e], c = col[e];
  float dx = pos[r * 3 + 0] - pos[c * 3 + 0];
  float dy = pos[r * 3 + 1] - pos[c * 3 + 1];
  float dz = pos[r * 3 + 2] - pos[c * 3 + 2];
  diff[(long)e * 3 + 0] = dx; diff[(long)e * 3 + 1] = dy; diff[(long)e * 3 + 2] = dz;
  dist[e] = fmaxf(sqrtf(dx * dx + dy * dy + dz * dz), 1e-5f);
}
__global__ void k_gauss(const float* dist, const float* cen, const float* wid, bf16* out, int E) {
  long t = (long)blockIdx.x * 256 + threadIdx.x;
  if (t >= (long)E * RR) return;
  int e = t >> 6, r = t & 63;
  float d = dist[e];
  float env = (d < CUTOFF) ? 0.5f * (__cosf(d * (3.14159265358979f / CUTOFF)) + 1.f) : 0.f;
  float u = (d - cen[r]) / (fabsf(wid[r]) + 1e-5f);
  out[t] = (bf16)(__expf(-0.5f * u * u) * env);
}
// rest[E,32]: col0 = dist, col1..13 = edge_attr, col14..31 = 0
__global__ void k_rest(const float* dist, const float* ea, bf16* out, int E) {
  long t = (long)blockIdx.x * 256 + threadIdx.x;
  if (t >= (long)E * 32) return;
  int e = t >> 5, j = t & 31;
  float v = (j == 0) ? dist[e] : (j <= BOND ? ea[(long)e * BOND + (j - 1)] : 0.f);
  out[t] = (bf16)v;
}
// edge_feat[E,96]: 0..12 = edge_attr, 13..76 = rbf_feat, 77..95 = 0
__global__ void k_edgefeat(const float* ea, const float* rbff, bf16* out, int E) {
  long t = (long)blockIdx.x * 256 + threadIdx.x;
  if (t >= (long)E * 96) return;
  long e = t / 96; int j = t % 96;
  float v = (j < BOND) ? ea[e * BOND + j] : (j < 77 ? rbff[e * RR + (j - BOND)] : 0.f);
  out[t] = (bf16)v;
}
__global__ void k_scatter_edge(const float* src, const float* pes, float cs,
                               const int* idx, float* out, int E, int D) {
  long t = (long)blockIdx.x * 256 + threadIdx.x;
  if (t >= (long)E * D) return;
  long e = t / D; int d = t % D;
  float v = src[t] * cs;
  if (pes) v *= pes[e];
  atomAdd(&out[(long)idx[e] * D + d], v);
}
__global__ void k_cf_scatter(const float* nx, const float* Wf, const int* row, const int* col,
                             float* agg, int E) {
  long t = (long)blockIdx.x * 256 + threadIdx.x;
  if (t >= (long)E * HH) return;
  long e = t >> 8; int d = t & 255;
  atomAdd(&agg[(long)row[e] * HH + d], nx[(long)col[e] * HH + d] * Wf[t]);
}
__global__ __launch_bounds__(32) void k_gat_alpha(const float* xl, const float* xr, const float* ee,
                                                  const float* attw, const int* srcI, const int* dstI,
                                                  float* alpha, float* mx, int E) {
  int e = blockIdx.x; if (e >= E) return;
  int l = threadIdx.x;
  int s = srcI[e], dd = dstI[e];
  float acc = 0.f;
  for (int j = 0; j < 8; j++) {
    int d = l * 8 + j;
    float v = xl[(long)s * HH + d] + xr[(long)dd * HH + d] + ee[(long)e * HH + d];
    v = v > 0.f ? v : 0.2f * v;
    acc += v * attw[d];
  }
  for (int sd = 1; sd < 8; sd <<= 1) acc += __shfl_xor(acc, sd, 32);
  if ((l & 7) == 0) {
    int h = l >> 3;
    alpha[(long)e * 4 + h] = acc;
    atomicMaxF(&mx[(long)dd * 4 + h], acc);
  }
}
__global__ void k_seg_max(const float* v, const int* idx, long n, int nh, float* mx) {
  long t = (long)blockIdx.x * 256 + threadIdx.x;
  if (t < n * nh) atomicMaxF(&mx[(long)idx[t / nh] * nh + t % nh], v[t]);
}
__global__ void k_seg_exp(const float* v, const int* idx, long n, int nh,
                          const float* mx, float* sm, float* a) {
  long t = (long)blockIdx.x * 256 + threadIdx.x;
  if (t >= n * nh) return;
  long s = (long)idx[t / nh] * nh + t % nh;
  float av = __expf(v[t] - mx[s]);
  a[t] = av;
  atomAdd(&sm[s], av);
}
__global__ void k_seg_div(float* a, const int* idx, long n, int nh, const float* sm) {
  long t = (long)blockIdx.x * 256 + threadIdx.x;
  if (t < n * nh) a[t] /= (sm[(long)idx[t / nh] * nh + t % nh] + 1e-16f);
}
__global__ void k_gat_scatter(const float* xl, const float* a, const int* srcI, const int* dstI,
                              float* out, int E) {
  long t = (long)blockIdx.x * 256 + threadIdx.x;
  if (t >= (long)E * HH) return;
  long e = t >> 8; int d = t & 255;
  atomAdd(&out[(long)dstI[e] * HH + d], xl[(long)srcI[e] * HH + d] * a[e * 4 + (d >> 6)]);
}
__global__ void k_gat_final(float* h, const float* gout, const float* bias, long n) {
  long t = (long)blockIdx.x * 256 + threadIdx.x;
  if (t < n * HH) h[t] += gout[t] + bias[t & 255];
}
__global__ void k_fb1(const float* dist, const float* w, const float* b, bf16* out, int E) {
  long t = (long)blockIdx.x * 256 + threadIdx.x;
  if (t >= (long)E * HH) return;
  long e = t >> 8; int j = t & 255;
  float z = dist[e] * w[j] + b[j];
  out[t] = (bf16)(z / (1.f + __expf(-z)));
}
__global__ void k_mul(const float* a, const float* b, float* o, long n) {
  long t = (long)blockIdx.x * 256 + threadIdx.x;
  if (t < n) o[t] = a[t] * b[t];
}
__global__ void k_add(float* a, const float* b, long n) {
  long t = (long)blockIdx.x * 256 + threadIdx.x;
  if (t < n) a[t] += b[t];
}
__global__ void k_addbc(float* h, const float* bc, const int* batch, long n) {
  long t = (long)blockIdx.x * 256 + threadIdx.x;
  if (t < n * HH) h[t] += bc[(long)batch[t >> 8] * HH + (t & 255)];
}
__global__ void k_scatter_node(const float* src, const float* pns, const int* idx,
                               float* out, long n, int D) {
  long t = (long)blockIdx.x * 256 + threadIdx.x;
  if (t >= n * D) return;
  long i = t / D; int d = t % D;
  float v = src[t];
  if (pns) v *= pns[i];
  atomAdd(&out[(long)idx[i] * D + d], v);
}
__global__ void k_count(const int* batch, float* cnt, long n) {
  long t = (long)blockIdx.x * 256 + threadIdx.x;
  if (t < n) atomAdd(&cnt[batch[t]], 1.f);
}
__global__ void k_vn_mean(float* aggB, const float* cnt, bf16* aggBb, int B) {
  long t = (long)blockIdx.x * 256 + threadIdx.x;
  if (t >= (long)B * HH) return;
  float c = fmaxf(cnt[t >> 8], 1.f);
  float v = aggB[t] / c;
  aggB[t] = v; aggBb[t] = (bf16)v;
}

// ---------------------------------------------------------------- host side
namespace {
struct Lin { const float* b; const float* w; int K, N, Kpad; bf16* wt; };
struct LNp { const float* b; const float* g; };
}

extern "C" void kernel_launch(void* const* d_in, const int* in_sizes, int n_in,
                              void* d_out, int out_size, void* d_ws, size_t ws_size,
                              hipStream_t stream) {
  const float* X   = (const float*)d_in[0];
  const int*   EI  = (const int*)d_in[1];
  const int*   ROW = EI;
  const int*   COL = EI + EE;
  const float* EA  = (const float*)d_in[2];
  const int*   BATCH = (const int*)d_in[3];
  const float* POS = (const float*)d_in[4];
  float* OUT = (float*)d_out;

  auto cdiv = [](long a, long b) { return (a + b - 1) / b; };

  // workspace bump allocator (256B aligned)
  char* wsB = (char*)d_ws;
  size_t off = 0;
  auto alloc = [&](size_t bytes) -> void* {
    void* p = wsB + off;
    off = (off + bytes + 255) & ~(size_t)255;
    return p;
  };

  // ---- parameter pytree walk (jax sorted-key flatten order), params start at d_in[5]
  int idx = 5;
  auto nextp = [&]() { return (const float*)d_in[idx++]; };
  auto getLinRaw = [&](int K, int N, bool bias) {
    Lin L; L.K = K; L.N = N; L.Kpad = (K + 31) & ~31;
    L.b = bias ? nextp() : nullptr; L.w = nextp(); L.wt = nullptr;
    return L;
  };
  auto getLin = [&](int K, int N, bool bias) {
    Lin L = getLinRaw(K, N, bias);
    L.wt = (bf16*)alloc((size_t)N * L.Kpad * sizeof(bf16));
    k_wt<<<cdiv((long)N * L.Kpad, 256), 256, 0, stream>>>(L.w, L.wt, K, N, L.Kpad);
    return L;
  };
  auto getLN = [&]() { LNp p; p.b = nextp(); p.g = nextp(); return p; };

  struct EgnnP { Lin att, c1; const float* c2w; Lin e1, e2; LNp e_ln; Lin fb1, fb2, n1, n2; LNp n_ln, norm; };
  struct GatP  { const float* attw; const float* bias; Lin lin_e, lin_l, lin_r; };
  struct CfP   { const float* cen; Lin f1, f2, f3, gate; LNp ln; Lin node, p1, p2; const float* wid; };
  struct VnP   { Lin a1, a2; LNp a_ln; Lin bc1; LNp bc_ln, norm; };
  struct BlkP  { bool hasCf; CfP cf; EgnnP eg; GatP gat; LNp norm; VnP vn; };
  struct RbfP  { const float* cen; Lin p1, p2; const float* wid; };

  auto getEgnn = [&]() {
    EgnnP g;
    g.att = getLinRaw(HH, 1, true);
    g.c1  = getLin(HH, HH, true);
    g.c2w = nextp();
    g.e1  = getLin(2 * HH + 1 + BOND, 2 * HH, true);   // Kpad = 544
    g.e2  = getLin(2 * HH, HH, true);
    g.e_ln = getLN();
    g.fb1 = getLinRaw(1, HH, true);
    g.fb2 = getLin(HH, HH, true);
    g.n1  = getLin(2 * HH, 2 * HH, true);
    g.n2  = getLin(2 * HH, HH, true);
    g.n_ln = getLN();
    g.norm = getLN();
    return g;
  };
  auto getGat = [&]() {
    GatP g; g.attw = nextp(); g.bias = nextp();
    g.lin_e = getLin(77, HH, true);                    // Kpad = 96
    g.lin_l = getLin(HH, HH, true); g.lin_r = getLin(HH, HH, true);
    return g;
  };
  auto getCf = [&]() {
    CfP c; c.cen = nextp();
    c.f1 = getLin(RR, HH, true); c.f2 = getLin(HH, HH, true); c.f3 = getLin(HH, HH, true);
    c.gate = getLin(HH, HH, true); c.ln = getLN(); c.node = getLin(HH, HH, true);
    c.p1 = getLin(RR, RR, true); c.p2 = getLin(RR, RR, true); c.wid = nextp();
    return c;
  };
  auto getVn = [&]() {
    VnP v; v.a1 = getLin(HH, 2 * HH, true); v.a2 = getLin(2 * HH, HH, true); v.a_ln = getLN();
    v.bc1 = getLin(HH, HH, true); v.bc_ln = getLN(); v.norm = getLN();
    return v;
  };
  auto getRbf = [&]() {
    RbfP r; r.cen = nextp(); r.p1 = getLin(RR, RR, true); r.p2 = getLin(RR, RR, true); r.wid = nextp();
    return r;
  };

  BlkP blk[4];
  for (int i = 0; i < 4; i++) {
    blk[i].hasCf = (i % 2 == 1);
    if (blk[i].hasCf) blk[i].cf = getCf();                    // odd: cf, egnn, gat, norm
    blk[i].eg = getEgnn();
    blk[i].gat = getGat();
    blk[i].norm = getLN();
    if (!blk[i].hasCf) blk[i].vn = getVn();                   // even: egnn, gat, norm, vn
  }
  RbfP edge_rbf = getRbf();
  Lin gate1 = getLin(HH, 128, true);
  Lin gate2 = getLinRaw(128, 1, true);
  Lin input1 = getLin(INF_, HH, true);                        // Kpad = 64
  LNp input_ln = getLN();
  Lin jk1 = getLin(4 * HH, 2 * HH, true);
  Lin jk2 = getLin(2 * HH, HH, true);
  LNp jk_ln = getLN();
  Lin out1 = getLin(HH, HH, true);
  LNp out_ln = getLN();
  Lin se1 = getLin(HH, RR, true);
  Lin se2 = getLin(RR, HH, true);

  // ---- activation buffers
  bf16*  xb     = (bf16*)alloc((size_t)NN * 64 * 2);          // x padded 47->64
  float* h      = (float*)alloc((size_t)NN * HH * 4);
  bf16*  h_b    = (bf16*)alloc((size_t)NN * HH * 2);
  bf16*  hcat   = (bf16*)alloc((size_t)NN * 4 * HH * 2);
  float* pos_c  = (float*)alloc((size_t)NN * 3 * 4);
  float* pd     = (float*)alloc((size_t)NN * 3 * 4);
  float* dist0  = (float*)alloc((size_t)EE * 4);
  float* distc  = (float*)alloc((size_t)EE * 4);
  float* diffc  = (float*)alloc((size_t)EE * 3 * 4);
  bf16*  gaussb = (bf16*)alloc((size_t)EE * RR * 2);
  float* zt     = (float*)alloc((size_t)EE * RR * 4);
  bf16*  ztb    = (bf16*)alloc((size_t)EE * RR * 2);
  float* r2f    = (float*)alloc((size_t)EE * RR * 4);
  bf16*  r2b    = (bf16*)alloc((size_t)EE * RR * 2);
  bf16*  efb    = (bf16*)alloc((size_t)EE * 96 * 2);          // edge_feat padded 77->96
  bf16*  restb  = (bf16*)alloc((size_t)EE * 32 * 2);          // [dist,ea] padded 14->32
  float* ebig   = (float*)alloc((size_t)EE * 512 * 4);
  bf16*  ebig_b = (bf16*)alloc((size_t)EE * 512 * 2);
  float* mbuf   = (float*)alloc((size_t)EE * HH * 4);
  bf16*  mbuf_b = (bf16*)alloc((size_t)EE * HH * 2);
  bf16*  fbz    = (bf16*)alloc((size_t)EE * HH * 2);
  float* attE   = (float*)alloc((size_t)EE * 4);
  float* cw     = (float*)alloc((size_t)EE * 4);
  float* alphaE = (float*)alloc((size_t)EE * 4 * 4);
  float* aE     = (float*)alloc((size_t)EE * 4 * 4);
  float* agg    = (float*)alloc((size_t)NN * HH * 4);
  bf16*  aggb   = (bf16*)alloc((size_t)NN * HH * 2);
  float* xl     = (float*)alloc((size_t)NN * HH * 4);
  float* xr     = (float*)alloc((size_t)NN * HH * 4);
  float* nbig   = (float*)alloc((size_t)NN * 512 * 4);
  bf16*  nbigb  = (bf16*)alloc((size_t)NN * 512 * 2);
  float* n2out  = (float*)alloc((size_t)NN * HH * 4);
  float* mx     = (float*)alloc((size_t)NN * 4 * 4);
  float* sm     = (float*)alloc((size_t)NN * 4 * 4);
  float* vnS    = (float*)alloc((size_t)BB * HH * 4);
  bf16*  vnb    = (bf16*)alloc((size_t)BB * HH * 2);
  float* aggB   = (float*)alloc((size_t)BB * HH * 4);
  bf16*  aggBb  = (bf16*)alloc((size_t)BB * HH * 2);
  float* z1     = (float*)alloc((size_t)BB * 512 * 4);
  bf16*  z1b    = (bf16*)alloc((size_t)BB * 512 * 2);
  float* z2     = (float*)alloc((size_t)BB * HH * 4);
  float* bcB    = (float*)alloc((size_t)BB * HH * 4);
  float* cnt    = (float*)alloc((size_t)BB * 4);
  float* gsc    = (float*)alloc((size_t)NN * 4);
  float* aN     = (float*)alloc((size_t)NN * 4);
  float* mxB    = (float*)alloc((size_t)BB * 4);
  float* smB    = (float*)alloc((size_t)BB * 4);
  float* pooled = (float*)alloc((size_t)BB * HH * 4);
  bf16*  pooledb= (bf16*)alloc((size_t)BB * HH * 2);
  float* outx   = (float*)alloc((size_t)BB * HH * 4);
  bf16*  outxb  = (bf16*)alloc((size_t)BB * HH * 2);
  bf16*  s1b    = (bf16*)alloc((size_t)BB * RR * 2);
  float* s2     = (float*)alloc((size_t)BB * HH * 4);

  const float NEGINF = -__builtin_huge_valf();

  // M % 16 == 0, Nout % 64 == 0, K % 32 == 0 for every call below.
  auto gemm = [&](const bf16* A, long lda, const int* gi, const bf16* wt, long ldw,
                  const float* bias, float* Cf, bf16* Cb, long ldc,
                  int M, int Nout, int K, int act, int accum) {
    dim3 g(M / 16, Nout / 64);
    k_gemm<<<g, 32, 0, stream>>>(A, lda, gi, wt, ldw, bias, Cf, Cb, ldc, Nout, K, act, accum);
  };
  auto ln = [&](const float* x, const float* res, const LNp& p, float* of, bf16* ob,
                long ldb, int D, long M) {
    k_ln<<<(unsigned)M, 256, 0, stream>>>(x, res, p.g, p.b, of, ob, ldb, D, M);
  };
  auto fill = [&](float* p, float v, long n) {
    k_fill<<<cdiv(n, 256), 256, 0, stream>>>(p, v, n);
  };
  auto f2b = [&](const float* s, bf16* d, long M, int Dsrc, int Dpad, long ldd) {
    k_f2b<<<cdiv(M * Dpad, 256), 256, 0, stream>>>(s, d, M, Dsrc, Dpad, ldd);
  };
  // rbf chain: gauss -> proj1(silu) -> proj2
  auto rbf_run = [&](const float* cen, const float* wid, const Lin& p1, const Lin& p2,
                     const float* dist, float* outF, bf16* outB) {
    k_gauss<<<cdiv((long)EE * RR, 256), 256, 0, stream>>>(dist, cen, wid, gaussb, EE);
    gemm(gaussb, RR, nullptr, p1.wt, p1.Kpad, p1.b, zt, ztb, RR, EE, RR, RR, 1, 0);
    gemm(ztb, RR, nullptr, p2.wt, p2.Kpad, p2.b, outF, outB, RR, EE, RR, RR, 0, 0);
  };

  // ================= forward =================
  hipMemcpyAsync(pos_c, POS, sizeof(float) * NN * 3, hipMemcpyDeviceToDevice, stream);
  k_edge_geom<<<cdiv(EE, 256), 256, 0, stream>>>(POS, ROW, COL, diffc, dist0, EE);
  rbf_run(edge_rbf.cen, edge_rbf.wid, edge_rbf.p1, edge_rbf.p2, dist0, r2f, nullptr);
  k_edgefeat<<<cdiv((long)EE * 96, 256), 256, 0, stream>>>(EA, r2f, efb, EE);

  f2b(X, xb, NN, INF_, 64, 64);
  gemm(xb, 64, nullptr, input1.wt, input1.Kpad, input1.b, agg, nullptr, HH, NN, HH, 64, 2, 0);
  ln(agg, nullptr, input_ln, h, h_b, HH, HH, NN);

  fill(vnS, 0.f, (long)BB * HH);
  fill(cnt, 0.f, BB);
  k_count<<<cdiv(NN, 256), 256, 0, stream>>>(BATCH, cnt, NN);

  for (int bi = 0; bi < 4; bi++) {
    const EgnnP& g = blk[bi].eg;
    const long ldE1 = g.e1.Kpad;   // 544
    // ---- EGNN
    k_edge_geom<<<cdiv(EE, 256), 256, 0, stream>>>(pos_c, ROW, COL, diffc, distc, EE);
    k_rest<<<cdiv((long)EE * 32, 256), 256, 0, stream>>>(distc, EA, restb, EE);
    gemm(h_b, HH, ROW, g.e1.wt, ldE1, nullptr, ebig, nullptr, 512, EE, 512, HH, 0, 0);
    gemm(h_b, HH, COL, g.e1.wt + 256, ldE1, nullptr, ebig, nullptr, 512, EE, 512, HH, 0, 1);
    gemm(restb, 32, nullptr, g.e1.wt + 512, ldE1, g.e1.b, ebig, nullptr, 512, EE, 512, 32, 1, 1);
    ln(ebig, nullptr, g.e_ln, nullptr, ebig_b, 512, 512, EE);
    gemm(ebig_b, 512, nullptr, g.e2.wt, 512, g.e2.b, mbuf, mbuf_b, HH, EE, HH, 512, 1, 0);
    k_rowdot<<<EE, 32, 0, stream>>>(mbuf, HH, g.att.w, g.att.b, attE, EE, 4);
    fill(agg, 0.f, (long)NN * HH);
    k_scatter_edge<<<cdiv((long)EE * HH, 256), 256, 0, stream>>>(mbuf, attE, 1.f, ROW, agg, EE, HH);
    f2b(agg, aggb, NN, HH, HH, HH);
    gemm(h_b, HH, nullptr, g.n1.wt, 512, nullptr, nbig, nullptr, 512, NN, 512, HH, 0, 0);
    gemm(aggb, HH, nullptr, g.n1.wt + 256, 512, g.n1.b, nbig, nullptr, 512, NN, 512, HH, 1, 1);
    ln(nbig, nullptr, g.n_ln, nullptr, nbigb, 512, 512, NN);
    gemm(nbigb, 512, nullptr, g.n2.wt, 512, g.n2.b, n2out, nullptr, HH, NN, HH, 512, 0, 0);
    ln(h, n2out, g.norm, h, h_b, HH, HH, NN);
    gemm(mbuf_b, HH, nullptr, g.c1.wt, 256, g.c1.b, ebig, nullptr, HH, EE, HH, HH, 1, 0);
    k_rowdot<<<EE, 32, 0, stream>>>(ebig, HH, g.c2w, nullptr, cw, EE, 0);
    fill(pd, 0.f, (long)NN * 3);
    k_scatter_edge<<<cdiv((long)EE * 3, 256), 256, 0, stream>>>(diffc, cw, 1.f, ROW, pd, EE, 3);
    k_add<<<cdiv((long)NN * 3, 256), 256, 0, stream>>>(pos_c, pd, (long)NN * 3);
    k_edge_geom<<<cdiv(EE, 256), 256, 0, stream>>>(pos_c, ROW, COL, diffc, distc, EE);
    k_fb1<<<cdiv((long)EE * HH, 256), 256, 0, stream>>>(distc, g.fb1.w, g.fb1.b, fbz, EE);
    gemm(fbz, HH, nullptr, g.fb2.wt, 256, g.fb2.b, ebig, nullptr, HH, EE, HH, HH, 0, 0);
    k_scatter_edge<<<cdiv((long)EE * HH, 256), 256, 0, stream>>>(ebig, nullptr, 0.1f, ROW, h, EE, HH);
    f2b(h, h_b, NN, HH, HH, HH);
    // ---- GAT
    const GatP& G = blk[bi].gat;
    gemm(h_b, HH, nullptr, G.lin_l.wt, 256, G.lin_l.b, xl, nullptr, HH, NN, HH, HH, 0, 0);
    gemm(h_b, HH, nullptr, G.lin_r.wt, 256, G.lin_r.b, xr, nullptr, HH, NN, HH, HH, 0, 0);
    gemm(efb, 96, nullptr, G.lin_e.wt, 96, G.lin_e.b, mbuf, nullptr, HH, EE, HH, 96, 0, 0);
    fill(mx, NEGINF, (long)NN * 4);
    k_gat_alpha<<<EE, 32, 0, stream>>>(xl, xr, mbuf, G.attw, ROW, COL, alphaE, mx, EE);
    fill(sm, 0.f, (long)NN * 4);
    k_seg_exp<<<cdiv((long)EE * 4, 256), 256, 0, stream>>>(alphaE, COL, EE, 4, mx, sm, aE);
    k_seg_div<<<cdiv((long)EE * 4, 256), 256, 0, stream>>>(aE, COL, EE, 4, sm);
    fill(agg, 0.f, (long)NN * HH);
    k_gat_scatter<<<cdiv((long)EE * HH, 256), 256, 0, stream>>>(xl, aE, ROW, COL, agg, EE);
    k_gat_final<<<cdiv((long)NN * HH, 256), 256, 0, stream>>>(h, agg, G.bias, NN);
    f2b(h, h_b, NN, HH, HH, HH);
    // ---- CFConv (odd blocks)
    if (blk[bi].hasCf) {
      const CfP& C = blk[bi].cf;
      rbf_run(C.cen, C.wid, C.p1, C.p2, dist0, r2f, r2b);
      gemm(r2b, RR, nullptr, C.f1.wt, 64, C.f1.b, nullptr, ebig_b, HH, EE, HH, RR, 1, 0);
      gemm(ebig_b, HH, nullptr, C.f2.wt, 256, C.f2.b, nullptr, mbuf_b, HH, EE, HH, HH, 1, 0);
      gemm(mbuf_b, HH, nullptr, C.f3.wt, 256, C.f3.b, ebig, nullptr, HH, EE, HH, HH, 0, 0);
      gemm(h_b, HH, nullptr, C.node.wt, 256, C.node.b, xl, nullptr, HH, NN, HH, HH, 0, 0);
      fill(agg, 0.f, (long)NN * HH);
      k_cf_scatter<<<cdiv((long)EE * HH, 256), 256, 0, stream>>>(xl, ebig, ROW, COL, agg, EE);
      ln(h, agg, C.ln, xr, aggb, HH, HH, NN);
      gemm(aggb, HH, nullptr, C.gate.wt, 256, C.gate.b, n2out, nullptr, HH, NN, HH, HH, 4, 0);
      k_mul<<<cdiv((long)NN * HH, 256), 256, 0, stream>>>(xr, n2out, h, (long)NN * HH);
    }
    // ---- VirtualNode (even blocks)
    if (!blk[bi].hasCf) {
      const VnP& V = blk[bi].vn;
      fill(aggB, 0.f, (long)BB * HH);
      k_scatter_node<<<cdiv((long)NN * HH, 256), 256, 0, stream>>>(h, nullptr, BATCH, aggB, NN, HH);
      k_vn_mean<<<cdiv((long)BB * HH, 256), 256, 0, stream>>>(aggB, cnt, aggBb, BB);
      gemm(aggBb, HH, nullptr, V.a1.wt, 256, V.a1.b, z1, nullptr, 512, BB, 512, HH, 2, 0);
      ln(z1, nullptr, V.a_ln, nullptr, z1b, 512, 512, BB);
      gemm(z1b, 512, nullptr, V.a2.wt, 512, V.a2.b, z2, nullptr, HH, BB, HH, 512, 0, 0);
      ln(vnS, z2, V.norm, vnS, vnb, HH, HH, BB);
      gemm(vnb, HH, nullptr, V.bc1.wt, 256, V.bc1.b, z2, nullptr, HH, BB, HH, HH, 2, 0);
      ln(z2, nullptr, V.bc_ln, bcB, nullptr, HH, HH, BB);
      k_addbc<<<cdiv((long)NN * HH, 256), 256, 0, stream>>>(h, bcB, BATCH, NN);
    }
    // ---- block norm + JK slice
    ln(h, nullptr, blk[bi].norm, h, h_b, HH, HH, NN);
    f2b(h, hcat + (long)bi * HH, NN, HH, HH, 4 * HH);
  }

  // ---- JK + readout
  gemm(hcat, 4 * HH, nullptr, jk1.wt, 1024, jk1.b, nbig, nullptr, 512, NN, 512, 4 * HH, 2, 0);
  ln(nbig, nullptr, jk_ln, nullptr, nbigb, 512, 512, NN);
  gemm(nbigb, 512, nullptr, jk2.wt, 512, jk2.b, xl, aggb, HH, NN, HH, 512, 0, 0);  // hfin=xl/aggb
  gemm(aggb, HH, nullptr, gate1.wt, 256, gate1.b, agg, nullptr, 128, NN, 128, HH, 3, 0);
  k_rowdot<<<NN, 32, 0, stream>>>(agg, 128, gate2.w, gate2.b, gsc, NN, 0);
  fill(mxB, NEGINF, BB);
  k_seg_max<<<cdiv(NN, 256), 256, 0, stream>>>(gsc, BATCH, NN, 1, mxB);
  fill(smB, 0.f, BB);
  k_seg_exp<<<cdiv(NN, 256), 256, 0, stream>>>(gsc, BATCH, NN, 1, mxB, smB, aN);
  k_seg_div<<<cdiv(NN, 256), 256, 0, stream>>>(aN, BATCH, NN, 1, smB);
  fill(pooled, 0.f, (long)BB * HH);
  k_scatter_node<<<cdiv((long)NN * HH, 256), 256, 0, stream>>>(xl, aN, BATCH, pooled, NN, HH);
  f2b(pooled, pooledb, BB, HH, HH, HH);
  gemm(pooledb, HH, nullptr, out1.wt, 256, out1.b, z1, nullptr, HH, BB, HH, HH, 2, 0);
  ln(z1, nullptr, out_ln, outx, outxb, HH, HH, BB);
  gemm(outxb, HH, nullptr, se1.wt, 256, se1.b, nullptr, s1b, RR, BB, RR, HH, 3, 0);
  gemm(s1b, RR, nullptr, se2.wt, 64, se2.b, s2, nullptr, HH, BB, HH, RR, 4, 0);
  k_mul<<<cdiv((long)BB * HH, 256), 256, 0, stream>>>(outx, s2, OUT, (long)BB * HH);

  (void)in_sizes; (void)n_in; (void)out_size; (void)ws_size;
}
